// GCN_PYG_84851373900205
// MI455X (gfx1250) — matrix operand order repaired and checked by reading, not compiled, not run on hardware
//
#include <hip/hip_runtime.h>
#include <hip/hip_bf16.h>
#include <math.h>

// ---------------------------------------------------------------------------
// GCN forward for MI455X (gfx1250). GEMMs use v_wmma_f32_16x16x32_bf16
// (f32 accumulate), edge aggregation uses wave32-per-edge with
// global_atomic_add_f32 (L2-resident: 25.6MB feature matrix << 192MB L2).
// ---------------------------------------------------------------------------

typedef __attribute__((ext_vector_type(16))) __bf16 v16bf;
typedef __attribute__((ext_vector_type(8)))  float  v8f;

union Frag32B { v16bf v; unsigned int u[8]; };

__device__ __forceinline__ unsigned short f2bf(float f) {
    unsigned int u = __float_as_uint(f);
    u += 0x7FFFu + ((u >> 16) & 1u);          // round-to-nearest-even
    return (unsigned short)(u >> 16);
}

__device__ __forceinline__ float elu1(float x) {
    return x > 0.0f ? x : (expf(x) - 1.0f);
}

// -------------------------- small utility kernels --------------------------

__global__ void k_fill(float* __restrict__ p, int n, float v) {
    int i = blockIdx.x * blockDim.x + threadIdx.x;
    if (i < n) p[i] = v;
}

__global__ void k_f32_to_bf16(const float* __restrict__ s,
                              unsigned short* __restrict__ d, int n) {
    int i = blockIdx.x * blockDim.x + threadIdx.x;
    if (i < n) d[i] = f2bf(s[i]);
}

// deg[col[e]] += w[e]   (self-loop "1.0" pre-filled by k_fill)
__global__ void k_deg(const int* __restrict__ col, const float* __restrict__ w,
                      float* __restrict__ deg, int E) {
    int i = blockIdx.x * blockDim.x + threadIdx.x;
    if (i < E) unsafeAtomicAdd(&deg[col[i]], w[i]);
}

__global__ void k_dinv(float* __restrict__ deg, int n) {
    int i = blockIdx.x * blockDim.x + threadIdx.x;
    if (i < n) {
        float d = deg[i];
        deg[i] = d > 0.0f ? rsqrtf(d) : 0.0f;
    }
}

// norm[e] = dinv[row]*w*dinv[col] for e<E ; dinv[i]^2 for self loops
__global__ void k_norm(const int* __restrict__ row, const int* __restrict__ col,
                       const float* __restrict__ w, const float* __restrict__ dinv,
                       float* __restrict__ norm, int E, int N) {
    int i = blockIdx.x * blockDim.x + threadIdx.x;
    if (i < E) {
        norm[i] = dinv[row[i]] * w[i] * dinv[col[i]];
    } else if (i < E + N) {
        float d = dinv[i - E];
        norm[i] = d * d;
    }
}

// ------------------------------ WMMA GEMM ----------------------------------
// Y[M,128] = X[M,128] @ W[128,128]^T   (W row n holds weights for out-col n)
// Wb is W pre-converted to bf16, row-major. mode==1: Y = elu(Y + bias).
// Block: 256 threads = 8 waves; block handles 16 rows x 128 cols,
// wave w owns column tile [16w,16w+16). K swept in 4 steps of 32.

#define ASTR 130   // LDS row stride in bf16 (65 dwords -> bank-conflict free)

__global__ void k_gemm_wmma(const float* __restrict__ X,
                            const unsigned short* __restrict__ Wb,
                            const float* __restrict__ bias,
                            float* __restrict__ Y, int M, int mode) {
    __shared__ unsigned short As[16 * ASTR];

    const int tid  = threadIdx.x;
    const int row0 = blockIdx.x * 16;

    // stage X tile (16x128) into LDS as bf16: 2048 elems / 256 threads = 8 each
    {
        int base = tid * 8;
        int r = base >> 7;
        int c = base & 127;
        const float* src = X + (size_t)(row0 + r) * 128 + c;
        bool ok = (row0 + r) < M;
#pragma unroll
        for (int j = 0; j < 8; ++j)
            As[r * ASTR + c + j] = ok ? f2bf(src[j]) : (unsigned short)0;
    }
    __syncthreads();

    const int wv   = tid >> 5;        // wave id 0..7 -> column tile
    const int lane = tid & 31;
    const int ll   = lane & 15;
    const int hi   = lane >> 4;
    const int col0 = wv * 16;

    v8f acc = {};
#pragma unroll
    for (int ks = 0; ks < 4; ++ks) {
        const int kbase = ks * 32;
        Frag32B A, B;
        // A fragment: row = ll ; VGPR i -> K = hi*16 + 2i
#pragma unroll
        for (int i = 0; i < 8; ++i) {
            int kk = kbase + hi * 16 + 2 * i;
            A.u[i] = *(const unsigned int*)&As[ll * ASTR + kk];
        }
        // B fragment: col (=W row) = col0+ll ; VGPR r -> K = hi*16 + 2r
#pragma unroll
        for (int r = 0; r < 8; ++r) {
            int kk = kbase + hi * 16 + 2 * r;
            B.u[r] = *(const unsigned int*)(Wb + (size_t)(col0 + ll) * 128 + kk);
        }
        acc = __builtin_amdgcn_wmma_f32_16x16x32_bf16(
            false, A.v, false, B.v, (short)0, acc, false, false);
    }

    // D layout: acc[v] -> row = row0 + hi*8 + v, col = col0 + ll
#pragma unroll
    for (int v = 0; v < 8; ++v) {
        int rr = row0 + hi * 8 + v;
        if (rr < M) {
            int cc = col0 + ll;
            float val = acc[v];
            if (mode) val = elu1(val + bias[cc]);
            Y[(size_t)rr * 128 + cc] = val;
        }
    }
}

// --------------------------- edge aggregation ------------------------------
// One wave32 per edge: O[col] += T[row] * norm  (128 floats, float4/lane)
__global__ void k_edge(const int* __restrict__ row, const int* __restrict__ col,
                       const float* __restrict__ norm,
                       const float* __restrict__ T, float* __restrict__ O,
                       int Etot, int E) {
    int w    = (blockIdx.x * blockDim.x + threadIdx.x) >> 5;
    int lane = threadIdx.x & 31;
    if (w >= Etot) return;
    int r, c;
    if (w < E) { r = row[w]; c = col[w]; }
    else       { r = c = w - E; }            // self loop
    float nv = norm[w];
    float4 t = *(const float4*)(T + (size_t)r * 128 + lane * 4);
    float* o = O + (size_t)c * 128 + lane * 4;
    unsafeAtomicAdd(o + 0, t.x * nv);
    unsafeAtomicAdd(o + 1, t.y * nv);
    unsafeAtomicAdd(o + 2, t.z * nv);
    unsafeAtomicAdd(o + 3, t.w * nv);
}

__global__ void k_bias_elu(float* __restrict__ Hb, const float* __restrict__ b, int n) {
    int i = blockIdx.x * blockDim.x + threadIdx.x;
    if (i < n) Hb[i] = elu1(Hb[i] + b[i & 127]);
}

// ------------------------------ mean pool ----------------------------------
__global__ void k_pool(const int* __restrict__ batch, const float* __restrict__ Hb,
                       float* __restrict__ pooled, float* __restrict__ cnt, int N) {
    int w    = (blockIdx.x * blockDim.x + threadIdx.x) >> 5;
    int lane = threadIdx.x & 31;
    if (w >= N) return;
    int b = batch[w];
    float4 h = *(const float4*)(Hb + (size_t)w * 128 + lane * 4);
    float* p = pooled + (size_t)b * 128 + lane * 4;
    unsafeAtomicAdd(p + 0, h.x);
    unsafeAtomicAdd(p + 1, h.y);
    unsafeAtomicAdd(p + 2, h.z);
    unsafeAtomicAdd(p + 3, h.w);
    if (lane == 0) unsafeAtomicAdd(&cnt[b], 1.0f);
}

__global__ void k_pooldiv(float* __restrict__ pooled, const float* __restrict__ cnt, int n) {
    int i = blockIdx.x * blockDim.x + threadIdx.x;
    if (i < n) pooled[i] = pooled[i] / fmaxf(cnt[i >> 7], 1.0f);
}

// ----------------------- fc2 + log_softmax (head) --------------------------
// One wave per graph: C dot-products of length 128 via butterfly reduce.
__global__ void k_head(const float* __restrict__ Z, const float* __restrict__ W2,
                       const float* __restrict__ b2, float* __restrict__ out,
                       int G, int C) {
    int g = blockIdx.x;
    int lane = threadIdx.x;
    if (g >= G) return;
    float4 z = *(const float4*)(Z + (size_t)g * 128 + lane * 4);
    float lg[16];
    for (int c = 0; c < C; ++c) {
        float4 wv = *(const float4*)(W2 + (size_t)c * 128 + lane * 4);
        float p = z.x * wv.x + z.y * wv.y + z.z * wv.z + z.w * wv.w;
#pragma unroll
        for (int off = 16; off; off >>= 1) p += __shfl_xor(p, off, 32);
        lg[c] = p + b2[c];
    }
    float m = lg[0];
    for (int c = 1; c < C; ++c) m = fmaxf(m, lg[c]);
    float s = 0.0f;
    for (int c = 0; c < C; ++c) s += expf(lg[c] - m);
    float l = m + logf(s);
    if (lane == 0)
        for (int c = 0; c < C; ++c) out[(size_t)g * C + c] = lg[c] - l;
}

// ------------------------------- launcher ----------------------------------

extern "C" void kernel_launch(void* const* d_in, const int* in_sizes, int n_in,
                              void* d_out, int out_size, void* d_ws, size_t ws_size,
                              hipStream_t stream) {
    const float* x     = (const float*)d_in[0];
    const int*   ei    = (const int*)d_in[1];
    const float* ew    = (const float*)d_in[2];
    const int*   batch = (const int*)d_in[3];
    const float* Wc[4] = { (const float*)d_in[4], (const float*)d_in[6],
                           (const float*)d_in[8], (const float*)d_in[10] };
    const float* bc[4] = { (const float*)d_in[5], (const float*)d_in[7],
                           (const float*)d_in[9], (const float*)d_in[11] };
    const float* fc1w  = (const float*)d_in[12];
    const float* fc1b  = (const float*)d_in[13];
    const float* fc2w  = (const float*)d_in[14];
    const float* fc2b  = (const float*)d_in[15];
    float* out = (float*)d_out;

    const int N  = in_sizes[3];          // 50000
    const int E  = in_sizes[2];          // 1600000
    const int Hh = in_sizes[5];          // 128
    const int Cc = in_sizes[15];         // 10
    const int Gg = out_size / Cc;        // 512
    const int ET = E + N;

    // workspace carve-out (256B aligned)
    char* ws = (char*)d_ws;
    size_t off = 0;
    auto carve = [&](size_t bytes) {
        void* p = ws + off;
        off += (bytes + 255) & ~(size_t)255;
        return p;
    };
    float*          deg    = (float*)carve((size_t)N * 4);
    float*          nrm    = (float*)carve((size_t)ET * 4);
    float*          T      = (float*)carve((size_t)N * Hh * 4);
    float*          Hb     = (float*)carve((size_t)N * Hh * 4);
    unsigned short* Wb     = (unsigned short*)carve((size_t)5 * Hh * Hh * 2);
    float*          pooled = (float*)carve((size_t)Gg * Hh * 4);
    float*          cnt    = (float*)carve((size_t)Gg * 4);
    float*          Z      = (float*)carve((size_t)Gg * Hh * 4);

    const int* row = ei;
    const int* col = ei + E;

    const int HH = Hh * Hh;                   // 16384
    // weights -> bf16
    for (int l = 0; l < 4; ++l)
        k_f32_to_bf16<<<(HH + 255) / 256, 256, 0, stream>>>(Wc[l], Wb + (size_t)l * HH, HH);
    k_f32_to_bf16<<<(HH + 255) / 256, 256, 0, stream>>>(fc1w, Wb + (size_t)4 * HH, HH);

    // gcn_norm
    k_fill<<<(N + 255) / 256, 256, 0, stream>>>(deg, N, 1.0f);   // self-loop weight
    k_deg <<<(E + 255) / 256, 256, 0, stream>>>(col, ew, deg, E);
    k_dinv<<<(N + 255) / 256, 256, 0, stream>>>(deg, N);
    k_norm<<<(ET + 255) / 256, 256, 0, stream>>>(row, col, ew, deg, nrm, E, N);

    // 4 GCN conv layers
    const float* hin = x;
    const int NH = N * Hh;
    for (int l = 0; l < 4; ++l) {
        k_gemm_wmma<<<(N + 15) / 16, 256, 0, stream>>>(hin, Wb + (size_t)l * HH, nullptr, T, N, 0);
        k_fill<<<(NH + 255) / 256, 256, 0, stream>>>(Hb, NH, 0.0f);
        k_edge<<<(ET + 7) / 8, 256, 0, stream>>>(row, col, nrm, T, Hb, ET, E);
        k_bias_elu<<<(NH + 255) / 256, 256, 0, stream>>>(Hb, bc[l], NH);
        hin = Hb;
    }

    // global mean pool
    k_fill<<<(Gg * Hh + 255) / 256, 256, 0, stream>>>(pooled, Gg * Hh, 0.0f);
    k_fill<<<(Gg + 255) / 256, 256, 0, stream>>>(cnt, Gg, 0.0f);
    k_pool<<<(N + 7) / 8, 256, 0, stream>>>(batch, Hb, pooled, cnt, N);
    k_pooldiv<<<(Gg * Hh + 255) / 256, 256, 0, stream>>>(pooled, cnt, Gg * Hh);

    // fc1 (+bias+ELU) via WMMA, then fc2 + log_softmax
    k_gemm_wmma<<<(Gg + 15) / 16, 256, 0, stream>>>(pooled, Wb + (size_t)4 * HH, fc1b, Z, Gg, 1);
    k_head<<<Gg, 32, 0, stream>>>(Z, fc2w, fc2b, out, Gg, Cc);
}
